// BigBirdAttention_39865886441714
// MI455X (gfx1250) — matrix operand order repaired and checked by reading, not compiled, not run on hardware
//
#include <hip/hip_runtime.h>

// ---------------------------------------------------------------------------
// BigBird attention for MI455X (gfx1250, wave32, WMMA).
// Precision strategy: bf16 operands, f32 WMMA accumulation everywhere
// (v_wmma_f32_16x16x32_bf16). Problem is matrix-op bound (~40 GFLOP vs ~50MB
// of traffic at 23.3 TB/s), so we feed the 16-bit WMMA pipe, and exploit
// BigBird sparsity (~1% of dense) via a per-Q-tile key-tile bitmask.
// GEMM stages the shared weight slab into LDS with async-to-LDS copies
// (ASYNCcnt double buffering) via the gfx1250 async builtins.
// ---------------------------------------------------------------------------

#define D_MODEL 1024
#define N_HEADS 16
#define HEAD_DIM 64
#define SEQ_L 2048
#define BATCH 2
#define MROWS (BATCH * SEQ_L) // 4096

typedef __attribute__((ext_vector_type(16))) __bf16 bf16x16;
typedef __attribute__((ext_vector_type(8)))  __bf16 bf16x8;
typedef __attribute__((ext_vector_type(8)))  float  f32x8;
typedef __attribute__((ext_vector_type(4)))  int    i32x4;

union Frag { bf16x16 f; bf16x8 h[2]; };

#if defined(__gfx1250__) && __has_builtin(__builtin_amdgcn_global_load_async_to_lds_b128) && \
    __has_builtin(__builtin_amdgcn_s_wait_asynccnt)
#define BB_ASYNC 1
#else
#define BB_ASYNC 0
#endif

#if BB_ASYNC
typedef __attribute__((address_space(1))) i32x4 g_i32x4; // global int4
typedef __attribute__((address_space(3))) i32x4 l_i32x4; // LDS int4
#endif

// 16-byte global -> LDS copy: async DMA (ASYNCcnt) when available.
static __device__ __forceinline__ void bb_cp16(void* lds_dst, const void* gsrc) {
#if BB_ASYNC
  __builtin_amdgcn_global_load_async_to_lds_b128(
      (g_i32x4*)(void*)gsrc,
      (l_i32x4*)lds_dst,
      0, 0);
#else
  *(bf16x8*)lds_dst = *(const bf16x8*)gsrc;
#endif
}

template <unsigned short N>
static __device__ __forceinline__ void bb_async_wait() {
#if BB_ASYNC
  __builtin_amdgcn_s_wait_asynccnt(N);
#endif
}

static __device__ __forceinline__ f32x8 wmma_bf16(const Frag& a, const Frag& b, f32x8 c) {
  // 8 args: (neg_a, A, neg_b, B, c_mod, C, reuse_a, reuse_b)
  return __builtin_amdgcn_wmma_f32_16x16x32_bf16(false, a.f, false, b.f, (short)0, c, false, false);
}

// Hash-based stand-in for jax uniform random column pick: floor(u*(i+1)).
static __device__ __forceinline__ int ridx_of(int i, int r) {
  unsigned h = (unsigned)i * 2654435761u ^ (0x9E3779B9u * (unsigned)(r + 1));
  h ^= h >> 16; h *= 0x85EBCA6Bu; h ^= h >> 13;
  float u = (float)(h & 0xFFFFFFu) * (1.0f / 16777216.0f);
  int v = (int)(u * (float)(i + 1));
  return v > i ? i : v;
}

// ------------------------------- f32 -> bf16 -------------------------------
__global__ void bb_cvt_f32_bf16(const float* __restrict__ s, __bf16* __restrict__ d, int n) {
  int i = blockIdx.x * blockDim.x + threadIdx.x;
  if (i < n) d[i] = (__bf16)s[i];
}

// ------------------------- generic NT GEMM (bf16) --------------------------
// C[M,N] = A[M,K] * W[N,K]^T, K = D_MODEL. Block = 128 thr (4 waves), tile
// 128x64; wave computes 32x64 (2x4 frags). W slab (64x32) is shared by all 4
// waves -> staged in LDS via double-buffered async copies.
// mode 0: bf16 row-major [M, D_MODEL]
// mode 1: bf16 V-transpose: out[(b*16+h)*64 + e][l]  (for attention B-frags)
// mode 2: f32 row-major (final projection -> d_out)
__global__ __launch_bounds__(128) void bb_gemm_nt(const __bf16* __restrict__ A,
                                                  const __bf16* __restrict__ W,
                                                  void* __restrict__ Cout, int mode) {
  __shared__ __align__(16) __bf16 wt[2][64][32];
  const int tid = threadIdx.x;
  const int lane = tid & 31;
  const int wave = tid >> 5;
  const int lm = lane & 15, sel = lane >> 4;
  const int m0 = blockIdx.x * 128 + wave * 32;
  const int n0 = blockIdx.y * 64;

  // Cooperative W staging: thread -> (row, 16-col half) of the 64x32 slab.
  const int crow = tid >> 1;
  const int ccol = (tid & 1) * 16;
  const __bf16* wsrc = W + (size_t)(n0 + crow) * D_MODEL + ccol;

  // Preload slab 0.
  bb_cp16(&wt[0][crow][ccol],     wsrc);
  bb_cp16(&wt[0][crow][ccol + 8], wsrc + 8);

  f32x8 acc[2][4] = {};
  for (int kc = 0; kc < D_MODEL; kc += 32) {
    const int cur = (kc >> 5) & 1;
    if (kc + 32 < D_MODEL) {
      const __bf16* ns = wsrc + kc + 32;
      bb_cp16(&wt[cur ^ 1][crow][ccol],     ns);
      bb_cp16(&wt[cur ^ 1][crow][ccol + 8], ns + 8);
      bb_async_wait<2>(); // slab `cur` (2 older transfers) complete, next in flight
    } else {
      bb_async_wait<0>();
    }
    __syncthreads(); // all waves' async writes to wt[cur] visible

    Frag a[2];
#pragma unroll
    for (int i = 0; i < 2; ++i) {
      // A frag, this code's contiguous k order (matches B): lane m in 0..15 holds K 0..15; lanes 16..31 hold K 16..31
      const __bf16* ap = A + (size_t)(m0 + i * 16 + lm) * D_MODEL + kc + sel * 16;
      a[i].h[0] = *(const bf16x8*)(ap);
      a[i].h[1] = *(const bf16x8*)(ap + 8);
      __builtin_prefetch(ap + 32, 0, 1); // next K slab -> global_prefetch_b8
    }
    Frag bw[4];
#pragma unroll
    for (int j = 0; j < 4; ++j) {
      // B frag, same contiguous k order as A: lane (n, kh): K = kh*16 + 0..15 of column n -> LDS row
      bw[j].h[0] = *(const bf16x8*)&wt[cur][j * 16 + lm][sel * 16];
      bw[j].h[1] = *(const bf16x8*)&wt[cur][j * 16 + lm][sel * 16 + 8];
    }
#pragma unroll
    for (int i = 0; i < 2; ++i)
#pragma unroll
      for (int j = 0; j < 4; ++j)
        acc[i][j] = wmma_bf16(a[i], bw[j], acc[i][j]);

    __syncthreads(); // done reading wt[cur] before it is refilled next round
  }

  // C layout: VGPR e -> row e (lanes 0-15) / row e+8 (lanes 16-31); N = lane%16
#pragma unroll
  for (int i = 0; i < 2; ++i)
#pragma unroll
    for (int j = 0; j < 4; ++j)
#pragma unroll
      for (int e = 0; e < 8; ++e) {
        int row = m0 + i * 16 + e + 8 * sel;
        int col = n0 + j * 16 + lm;
        float v = acc[i][j][e];
        if (mode == 0) {
          ((__bf16*)Cout)[(size_t)row * D_MODEL + col] = (__bf16)v;
        } else if (mode == 1) {
          int bb = row >> 11, l = row & (SEQ_L - 1);
          int hh = col >> 6, eh = col & 63;
          ((__bf16*)Cout)[((size_t)((bb * N_HEADS + hh) * HEAD_DIM + eh)) * SEQ_L + l] = (__bf16)v;
        } else {
          ((float*)Cout)[(size_t)row * D_MODEL + col] = v;
        }
      }
}

// --------------------------------- RoPE ------------------------------------
__global__ void bb_rope(__bf16* __restrict__ qb, __bf16* __restrict__ kb,
                        const float* __restrict__ freqs) {
  int idx = blockIdx.x * blockDim.x + threadIdx.x;
  const int total = MROWS * N_HEADS * (HEAD_DIM / 2);
  if (idx >= total) return;
  int j = idx & 31;
  int h = (idx >> 5) & (N_HEADS - 1);
  int t = idx >> 9;            // 0..MROWS-1
  int l = t & (SEQ_L - 1);     // sequence position
  float f = freqs[l * 32 + j];
  float c = __cosf(f), s2 = __sinf(f);
  size_t off = (size_t)t * D_MODEL + h * HEAD_DIM + 2 * j;
  float re = (float)qb[off], im = (float)qb[off + 1];
  qb[off]     = (__bf16)(re * c - im * s2);
  qb[off + 1] = (__bf16)(re * s2 + im * c);
  re = (float)kb[off]; im = (float)kb[off + 1];
  kb[off]     = (__bf16)(re * c - im * s2);
  kb[off + 1] = (__bf16)(re * s2 + im * c);
}

// --------------------- BigBird flash attention (1 wave) --------------------
__global__ __launch_bounds__(32) void bb_attn(const __bf16* __restrict__ q,
                                              const __bf16* __restrict__ k,
                                              const __bf16* __restrict__ vT,
                                              __bf16* __restrict__ ctx) {
  __shared__ unsigned tmask[2];
  __shared__ __align__(16) __bf16 pst[16][32]; // P tile: C-layout -> A-layout relay
  const int lane = threadIdx.x;
  const int lm = lane & 15, sel = lane >> 4;
  const int nqt = SEQ_L / 16;
  const int qt = blockIdx.x % nqt;
  const int bh = blockIdx.x / nqt;
  const int h = bh & (N_HEADS - 1), b = bh / N_HEADS;
  const int q0 = qt * 16;

  // Build 64-bit allowed-key-tile mask (32 keys/tile). Row 0 attends all.
  if (lane < 2) tmask[lane] = (q0 == 0) ? 0xFFFFFFFFu : 0u;
  __syncthreads();
  if (q0 != 0 && lane < 16) {
    int i = q0 + lane;
    unsigned long long mb = 1ull;                     // tile 0: global column
    int lo = (i - 7) >> 5; if (lo < 0) lo = 0;
    int hi = i >> 5;                                  // causal window [i-7, i]
    for (int t = lo; t <= hi; ++t) mb |= (1ull << t);
    mb |= (1ull << (ridx_of(i, 0) >> 5));
    mb |= (1ull << (ridx_of(i, 1) >> 5));
    atomicOr(&tmask[0], (unsigned)mb);
    atomicOr(&tmask[1], (unsigned)(mb >> 32));
  }
  __syncthreads();
  const unsigned tm0 = tmask[0], tm1 = tmask[1];

  // Q A-fragments for hd chunks [0,32) and [32,64)
  Frag aq[2];
  const __bf16* qrow = q + (size_t)(b * SEQ_L + q0 + lm) * D_MODEL + h * HEAD_DIM;
  aq[0].h[0] = *(const bf16x8*)(qrow + sel * 16);
  aq[0].h[1] = *(const bf16x8*)(qrow + 8 + sel * 16);
  aq[1].h[0] = *(const bf16x8*)(qrow + 32 + sel * 16);
  aq[1].h[1] = *(const bf16x8*)(qrow + 40 + sel * 16);

  int r0[8], r1[8];
  float mrow[8], lrow[8];
#pragma unroll
  for (int e = 0; e < 8; ++e) {
    int i = q0 + e + 8 * sel;
    r0[e] = ridx_of(i, 0); r1[e] = ridx_of(i, 1);
    mrow[e] = -3.0e38f; lrow[e] = 0.0f;
  }
  f32x8 accv[4] = {};
  const __bf16* vbase = vT + (size_t)bh * HEAD_DIM * SEQ_L;

  for (int kt = 0; kt < SEQ_L / 32; ++kt) {
    unsigned bit = (kt < 32) ? ((tm0 >> kt) & 1u) : ((tm1 >> (kt - 32)) & 1u);
    if (!bit) continue;
    const int kb = kt * 32;

    // scores: 16x32 = two 16x16 WMMA outputs, each accumulated over hd=64
    f32x8 s[2];
#pragma unroll
    for (int st = 0; st < 2; ++st) {
      Frag b0, b1;
      const __bf16* krow = k + (size_t)(b * SEQ_L + kb + st * 16 + lm) * D_MODEL + h * HEAD_DIM + sel * 16;
      b0.h[0] = *(const bf16x8*)(krow);       b0.h[1] = *(const bf16x8*)(krow + 8);
      b1.h[0] = *(const bf16x8*)(krow + 32);  b1.h[1] = *(const bf16x8*)(krow + 40);
      f32x8 z = {};
      s[st] = wmma_bf16(aq[0], b0, z);
      s[st] = wmma_bf16(aq[1], b1, s[st]);
    }

    // scale + BigBird mask, then online softmax
    float tmax[8];
#pragma unroll
    for (int e = 0; e < 8; ++e) {
      int i = q0 + e + 8 * sel;
#pragma unroll
      for (int st = 0; st < 2; ++st) {
        int j = kb + st * 16 + lm;
        float v = s[st][e] * 0.125f; // 1/sqrt(64)
        bool causal = (j <= i);
        bool allow = (i == 0) || (j == 0) ||
                     (causal && ((i - j <= 7) || (j == r0[e]) || (j == r1[e]) || (i < 2)));
        s[st][e] = allow ? v : -3.0e38f;
      }
      tmax[e] = fmaxf(s[0][e], s[1][e]);
    }
#pragma unroll
    for (int off = 1; off < 16; off <<= 1)
#pragma unroll
      for (int e = 0; e < 8; ++e)
        tmax[e] = fmaxf(tmax[e], __shfl_xor(tmax[e], off, 32));

    float corr[8], rs[8];
#pragma unroll
    for (int e = 0; e < 8; ++e) {
      float mnew = fmaxf(mrow[e], tmax[e]);
      corr[e] = __expf(mrow[e] - mnew);
      mrow[e] = mnew;
      float p0 = __expf(s[0][e] - mnew);
      float p1 = __expf(s[1][e] - mnew);
      s[0][e] = p0; s[1][e] = p1;
      rs[e] = p0 + p1;
    }
#pragma unroll
    for (int off = 1; off < 16; off <<= 1)
#pragma unroll
      for (int e = 0; e < 8; ++e)
        rs[e] += __shfl_xor(rs[e], off, 32);
#pragma unroll
    for (int e = 0; e < 8; ++e) lrow[e] = lrow[e] * corr[e] + rs[e];
#pragma unroll
    for (int jf = 0; jf < 4; ++jf)
#pragma unroll
      for (int e = 0; e < 8; ++e) accv[jf][e] *= corr[e];

    // P: C-layout -> A-layout via LDS
    __syncthreads();
#pragma unroll
    for (int st = 0; st < 2; ++st)
#pragma unroll
      for (int e = 0; e < 8; ++e)
        pst[e + 8 * sel][st * 16 + lm] = (__bf16)s[st][e];
    __syncthreads();
    Frag pa;
    pa.h[0] = *(const bf16x8*)&pst[lm][sel * 16];
    pa.h[1] = *(const bf16x8*)&pst[lm][sel * 16 + 8];

    // out += P(16x32) @ V(32x64): V^T rows are contiguous -> direct B-frags
#pragma unroll
    for (int jf = 0; jf < 4; ++jf) {
      Frag bv;
      const __bf16* vp = vbase + (size_t)(jf * 16 + lm) * SEQ_L + kb + sel * 16;
      bv.h[0] = *(const bf16x8*)(vp);
      bv.h[1] = *(const bf16x8*)(vp + 8);
      accv[jf] = wmma_bf16(pa, bv, accv[jf]);
    }
  }

  float inv[8];
#pragma unroll
  for (int e = 0; e < 8; ++e) inv[e] = 1.0f / lrow[e];
#pragma unroll
  for (int jf = 0; jf < 4; ++jf)
#pragma unroll
    for (int e = 0; e < 8; ++e)
      ctx[(size_t)(b * SEQ_L + q0 + e + 8 * sel) * D_MODEL + h * HEAD_DIM + jf * 16 + lm] =
          (__bf16)(accv[jf][e] * inv[e]);
}

// ------------------------------- launcher ----------------------------------
extern "C" void kernel_launch(void* const* d_in, const int* in_sizes, int n_in,
                              void* d_out, int out_size, void* d_ws, size_t ws_size,
                              hipStream_t stream) {
  const float* x  = (const float*)d_in[0];
  const float* fr = (const float*)d_in[1];
  const float* wq = (const float*)d_in[2];
  const float* wk = (const float*)d_in[3];
  const float* wv = (const float*)d_in[4];
  const float* wo = (const float*)d_in[5];

  char* ws = (char*)d_ws;
  size_t off = 0;
  const size_t xbytes = (size_t)MROWS * D_MODEL * sizeof(__bf16);
  const size_t wbytes = (size_t)D_MODEL * D_MODEL * sizeof(__bf16);
  __bf16* xb  = (__bf16*)(ws + off); off += xbytes;
  __bf16* wqb = (__bf16*)(ws + off); off += wbytes;
  __bf16* wkb = (__bf16*)(ws + off); off += wbytes;
  __bf16* wvb = (__bf16*)(ws + off); off += wbytes;
  __bf16* wob = (__bf16*)(ws + off); off += wbytes;
  __bf16* qb  = (__bf16*)(ws + off); off += xbytes;
  __bf16* kb  = (__bf16*)(ws + off); off += xbytes;
  __bf16* vtb = (__bf16*)(ws + off); off += xbytes;
  __bf16* ctx = (__bf16*)(ws + off); off += xbytes;

  const int nx = MROWS * D_MODEL;
  const int nw = D_MODEL * D_MODEL;
  bb_cvt_f32_bf16<<<(nx + 255) / 256, 256, 0, stream>>>(x,  xb,  nx);
  bb_cvt_f32_bf16<<<(nw + 255) / 256, 256, 0, stream>>>(wq, wqb, nw);
  bb_cvt_f32_bf16<<<(nw + 255) / 256, 256, 0, stream>>>(wk, wkb, nw);
  bb_cvt_f32_bf16<<<(nw + 255) / 256, 256, 0, stream>>>(wv, wvb, nw);
  bb_cvt_f32_bf16<<<(nw + 255) / 256, 256, 0, stream>>>(wo, wob, nw);

  dim3 gg(MROWS / 128, D_MODEL / 64);
  bb_gemm_nt<<<gg, 128, 0, stream>>>(xb, wqb, (void*)qb,  0);
  bb_gemm_nt<<<gg, 128, 0, stream>>>(xb, wkb, (void*)kb,  0);
  bb_gemm_nt<<<gg, 128, 0, stream>>>(xb, wvb, (void*)vtb, 1);

  const int nrope = MROWS * N_HEADS * (HEAD_DIM / 2);
  bb_rope<<<(nrope + 255) / 256, 256, 0, stream>>>(qb, kb, fr);

  bb_attn<<<BATCH * N_HEADS * (SEQ_L / 16), 32, 0, stream>>>(qb, kb, vtb, ctx);

  bb_gemm_nt<<<gg, 128, 0, stream>>>(ctx, wob, d_out, 2);
}
